// NPUQwen3VLMoeTextExperts_63161789055057
// MI455X (gfx1250) — matrix operand order repaired and checked by reading, not compiled, not run on hardware
//
#include <hip/hip_runtime.h>

// Problem constants (from reference)
#define NUM_EXPERTS 16
#define HIDDEN      2048
#define INTER       768
#define TOPK        8
#define NTOK        4096

// Tiling
#define MT 128       // token rows per block
#define KT 128       // K chunk staged in LDS
#define LD 136       // LDS row stride in ushorts (272B rows, 16B aligned)

typedef __bf16 bf16_t;
typedef bf16_t v16bf __attribute__((ext_vector_type(16)));
typedef float  v8f   __attribute__((ext_vector_type(8)));

struct alignas(16) U128 { unsigned int a, b, c, d; };
struct alignas(8)  U64  { unsigned int lo, hi; };

__device__ __forceinline__ unsigned short f2bf_raw(float f) {
    return __builtin_bit_cast(unsigned short, (__bf16)f);
}
// Pack two floats to bf16x2 (native cvt, RNE)
__device__ __forceinline__ unsigned int pack2bf(float lo, float hi) {
    return (unsigned int)f2bf_raw(lo) | ((unsigned int)f2bf_raw(hi) << 16);
}

// A fragment (16x32 bf16): lane<16 row=lane holds K {0..7, 16..23}; lane>=16
// row=lane-16 holds K {8..15, 24..31}. Caller passes &lds[row][k2 + kbase]
// where kbase = (lane>=16)*8; halves 0..7 contiguous, halves 8..15 at +16.
__device__ __forceinline__ v16bf frag_a(const unsigned short* p) {
    union { U128 q[2]; v16bf v; } u;
    u.q[0] = *reinterpret_cast<const U128*>(p);
    u.q[1] = *reinterpret_cast<const U128*>(p + 16);
    return u.v;
}

// B fragment (32x16 bf16): lane<16 col=lane holds K {0..7, 16..23}; lane>=16
// col=lane-16 holds K {8..15, 24..31}. With LDS staged transposed [col][k], lane
// reads &lds[col][k2 + koff], koff = (lane>=16)*8; halves 0..7 and at +16.
__device__ __forceinline__ v16bf frag_b(const unsigned short* p) {
    union { U128 q[2]; v16bf v; } u;
    u.q[0] = *reinterpret_cast<const U128*>(p);
    u.q[1] = *reinterpret_cast<const U128*>(p + 16);
    return u.v;
}

#define WMMA_BF16(A, B, C) \
    __builtin_amdgcn_wmma_f32_16x16x32_bf16(false, (A), false, (B), (short)0, (C), false, false)

// combine[t, e] = sum_k routing_weights[t, idx] where router_indices[t,k]==e
__global__ __launch_bounds__(256)
void build_combine(const float* __restrict__ rw, const int* __restrict__ ridx,
                   float* __restrict__ combine) {
    int t = blockIdx.x * blockDim.x + threadIdx.x;
    if (t >= NTOK) return;
    int id[TOPK];
#pragma unroll
    for (int k = 0; k < TOPK; ++k) id[k] = ridx[t * TOPK + k];
#pragma unroll
    for (int e = 0; e < NUM_EXPERTS; ++e) {
        float w = rw[t * NUM_EXPERTS + e];
        float s = 0.0f;
#pragma unroll
        for (int k = 0; k < TOPK; ++k)
            if (id[k] == e) s += w;
        combine[t * NUM_EXPERTS + e] = s;
    }
}

__global__ __launch_bounds__(256)
void zero_out_kernel(float* __restrict__ out, int n) {
    int i = blockIdx.x * blockDim.x + threadIdx.x;
    if (i < n) out[i] = 0.0f;
}

// Pass 1: inter[z][N, 768] = combine[:, e] * silu(X @ Wg[e]) * (X @ Wu[e]),
// e = expertBase + blockIdx.z, stored bf16 (combine weight folded in here so
// pass 2 can accumulate experts directly; valid since down-proj is linear).
// Block: 128 tokens x (64 gate + 64 up cols). 8 waves in 4(row)x2(col) grid;
// each wave: 32 rows x (32 gate + 32 up) = 2 A frags, 4 B frags, 8 WMMA/k-step.
__global__ __launch_bounds__(256)
void moe_gateup_swiglu(const float* __restrict__ X,        // [NTOK][HIDDEN]
                       const float* __restrict__ GU,       // [E][HIDDEN][2*INTER]
                       const float* __restrict__ combine,  // [NTOK][NUM_EXPERTS]
                       unsigned short* __restrict__ interOut, // [gridDim.z][NTOK][INTER] bf16
                       int expertBase)
{
    __shared__ __align__(16) unsigned short Xs[MT][LD];   // [row][k]
    __shared__ __align__(16) unsigned short Ws[128][LD];  // [col][k]; col<64 gate, col>=64 up

    const int tid  = threadIdx.x;
    const int lane = tid & 31;
    const int wave = tid >> 5;
    const int m0 = blockIdx.y * MT;
    const int f0 = blockIdx.x * 64;
    const int e  = expertBase + blockIdx.z;
    const float* W = GU + (size_t)e * HIDDEN * (2 * INTER);
    unsigned short* io = interOut + (size_t)blockIdx.z * NTOK * INTER;

    const int rw = wave & 3;      // row strip (32 rows)
    const int cw = wave >> 2;     // col strip (32 gate + 32 up cols)
    const int r0 = rw * 32;
    const int cgb = cw * 32;      // gate col base in Ws; up at 64 + cgb

    v8f accg[2][2], accu[2][2];
#pragma unroll
    for (int i = 0; i < 2; ++i)
#pragma unroll
        for (int t = 0; t < 2; ++t) { accg[i][t] = v8f{}; accu[i][t] = v8f{}; }

    const int am    = r0 + (lane & 15);
    const int akb   = (lane >> 4) * 8;
    const int bkoff = (lane >> 4) * 8;
    const int bn    = lane & 15;

    for (int kk = 0; kk < HIDDEN; kk += KT) {
        // Stage X tile [MT][KT] as bf16: float4 loads, b64 LDS stores
#pragma unroll 4
        for (int i = 0; i < (MT * KT) / (256 * 4); ++i) {
            int idx = (i * 256 + tid) * 4;
            int r = idx >> 7;          // / KT
            int c = idx & (KT - 1);
            float4 v = *reinterpret_cast<const float4*>(
                &X[(size_t)(m0 + r) * HIDDEN + kk + c]);
            U64 p{pack2bf(v.x, v.y), pack2bf(v.z, v.w)};
            *reinterpret_cast<U64*>(&Xs[r][c]) = p;
        }
        // Stage W tile transposed [col][k]: 4 k-values per thread, b64 store
#pragma unroll 4
        for (int i = 0; i < (KT * 128) / (256 * 4); ++i) {
            int idx = i * 256 + tid;
            int c = idx & 127;
            int k = (idx >> 7) * 4;
            int gc = f0 + c + ((c >> 6) * (INTER - 64)); // c>=64 -> up half
            const float* wp = &W[(size_t)(kk + k) * (2 * INTER) + gc];
            U64 p{pack2bf(wp[0], wp[2 * INTER]),
                  pack2bf(wp[2 * (size_t)(2 * INTER)], wp[3 * (size_t)(2 * INTER)])};
            *reinterpret_cast<U64*>(&Ws[c][k]) = p;
        }
        __syncthreads();

        // Prefetch next chunk while the MMA pipe is busy
        if (kk + KT < HIDDEN) {
            int r = (tid * 4) >> 7, c = (tid * 4) & (KT - 1);
            __builtin_prefetch(&X[(size_t)(m0 + r) * HIDDEN + kk + KT + c], 0, 0);
            __builtin_prefetch(&W[(size_t)(kk + KT + (tid >> 7)) * (2 * INTER) + f0 + (tid & 127)], 0, 0);
        }

#pragma unroll
        for (int k2 = 0; k2 < KT; k2 += 32) {
            v16bf a0  = frag_a(&Xs[am][k2 + akb]);
            v16bf a1  = frag_a(&Xs[am + 16][k2 + akb]);
            v16bf bg0 = frag_b(&Ws[cgb + bn][k2 + bkoff]);
            v16bf bg1 = frag_b(&Ws[cgb + 16 + bn][k2 + bkoff]);
            v16bf bu0 = frag_b(&Ws[64 + cgb + bn][k2 + bkoff]);
            v16bf bu1 = frag_b(&Ws[64 + cgb + 16 + bn][k2 + bkoff]);
            accg[0][0] = WMMA_BF16(a0, bg0, accg[0][0]);
            accg[0][1] = WMMA_BF16(a0, bg1, accg[0][1]);
            accu[0][0] = WMMA_BF16(a0, bu0, accu[0][0]);
            accu[0][1] = WMMA_BF16(a0, bu1, accu[0][1]);
            accg[1][0] = WMMA_BF16(a1, bg0, accg[1][0]);
            accg[1][1] = WMMA_BF16(a1, bg1, accg[1][1]);
            accu[1][0] = WMMA_BF16(a1, bu0, accu[1][0]);
            accu[1][1] = WMMA_BF16(a1, bu1, accu[1][1]);
        }
        __syncthreads();
    }

    // SwiGLU * combine: gate & up fragments share the exact lane layout.
    // D layout: VGPR r -> row = r + (lane>=16 ? 8 : 0), col = lane&15.
    const int rb = m0 + r0 + ((lane >> 4) << 3);
    const int cb = f0 + cgb + (lane & 15);
#pragma unroll
    for (int i = 0; i < 2; ++i) {
#pragma unroll
        for (int r = 0; r < 8; ++r) {
            int row = rb + i * 16 + r;
            float cwgt = combine[(size_t)row * NUM_EXPERTS + e];
#pragma unroll
            for (int t = 0; t < 2; ++t) {
                float g = accg[i][t][r];
                float u = accu[i][t][r];
                float s = g / (1.0f + __expf(-g));
                io[(size_t)row * INTER + cb + t * 16] = f2bf_raw(cwgt * s * u);
            }
        }
    }
}

// Pass 2: out[N, HIDDEN] (+)= sum over experts of inter[e] @ W2[e].
// Combine weights already folded into inter. Each block loops expertCount
// experts with f32 accumulators held in registers (merged path: one launch,
// out written exactly once; fallback: expertCount==1 and accumulate!=0).
// Block: 128 tokens x 128 hidden cols. 8 waves in 4x2; each wave 32x64.
__global__ __launch_bounds__(256)
void moe_down_combine(const unsigned short* __restrict__ inter, // [EC][NTOK][INTER] bf16
                      const float* __restrict__ DW,             // [E][INTER][HIDDEN]
                      float* __restrict__ out,                  // [NTOK][HIDDEN]
                      int expertBase, int expertCount, int accumulate)
{
    __shared__ __align__(16) unsigned short As[MT][LD];   // [row][k]
    __shared__ __align__(16) unsigned short Bs[128][LD];  // [col][k]

    const int tid  = threadIdx.x;
    const int lane = tid & 31;
    const int wave = tid >> 5;
    const int m0 = blockIdx.y * MT;
    const int h0 = blockIdx.x * 128;

    const int rw = wave & 3;
    const int cw = wave >> 2;
    const int r0 = rw * 32;
    const int cb0 = cw * 64;

    v8f acc[2][4];
#pragma unroll
    for (int i = 0; i < 2; ++i)
#pragma unroll
        for (int t = 0; t < 4; ++t) acc[i][t] = v8f{};

    const int am    = r0 + (lane & 15);
    const int akb   = (lane >> 4) * 8;
    const int bkoff = (lane >> 4) * 8;
    const int bn    = lane & 15;

    for (int ei = 0; ei < expertCount; ++ei) {
        const unsigned short* A = inter + (size_t)ei * NTOK * INTER;
        const float* W = DW + (size_t)(expertBase + ei) * INTER * HIDDEN;

        for (int kk = 0; kk < INTER; kk += KT) {
            // Stage A tile: straight bf16 copy, b64 loads + b64 LDS stores
#pragma unroll 4
            for (int i = 0; i < (MT * KT) / (256 * 4); ++i) {
                int idx = (i * 256 + tid) * 4;
                int r = idx >> 7;
                int c = idx & (KT - 1);
                *reinterpret_cast<U64*>(&As[r][c]) =
                    *reinterpret_cast<const U64*>(&A[(size_t)(m0 + r) * INTER + kk + c]);
            }
            // Stage W tile transposed [col][k]: f32 -> bf16
#pragma unroll 4
            for (int i = 0; i < (KT * 128) / (256 * 4); ++i) {
                int idx = i * 256 + tid;
                int c = idx & 127;
                int k = (idx >> 7) * 4;
                const float* wp = &W[(size_t)(kk + k) * HIDDEN + h0 + c];
                U64 p{pack2bf(wp[0], wp[HIDDEN]),
                      pack2bf(wp[2 * (size_t)HIDDEN], wp[3 * (size_t)HIDDEN])};
                *reinterpret_cast<U64*>(&Bs[c][k]) = p;
            }
            __syncthreads();

            if (kk + KT < INTER) {
                int r = (tid * 4) >> 7, c = (tid * 4) & (KT - 1);
                __builtin_prefetch(&A[(size_t)(m0 + r) * INTER + kk + KT + c], 0, 0);
                __builtin_prefetch(&W[(size_t)(kk + KT + (tid >> 7)) * HIDDEN + h0 + (tid & 127)], 0, 0);
            }

#pragma unroll
            for (int k2 = 0; k2 < KT; k2 += 32) {
                v16bf a0 = frag_a(&As[am][k2 + akb]);
                v16bf a1 = frag_a(&As[am + 16][k2 + akb]);
                v16bf b0 = frag_b(&Bs[cb0 + bn][k2 + bkoff]);
                v16bf b1 = frag_b(&Bs[cb0 + 16 + bn][k2 + bkoff]);
                v16bf b2 = frag_b(&Bs[cb0 + 32 + bn][k2 + bkoff]);
                v16bf b3 = frag_b(&Bs[cb0 + 48 + bn][k2 + bkoff]);
                acc[0][0] = WMMA_BF16(a0, b0, acc[0][0]);
                acc[0][1] = WMMA_BF16(a0, b1, acc[0][1]);
                acc[0][2] = WMMA_BF16(a0, b2, acc[0][2]);
                acc[0][3] = WMMA_BF16(a0, b3, acc[0][3]);
                acc[1][0] = WMMA_BF16(a1, b0, acc[1][0]);
                acc[1][1] = WMMA_BF16(a1, b1, acc[1][1]);
                acc[1][2] = WMMA_BF16(a1, b2, acc[1][2]);
                acc[1][3] = WMMA_BF16(a1, b3, acc[1][3]);
            }
            __syncthreads();
        }
    }

    const int rb  = m0 + r0 + ((lane >> 4) << 3);
    const int cbl = h0 + cb0 + (lane & 15);
#pragma unroll
    for (int i = 0; i < 2; ++i) {
#pragma unroll
        for (int r = 0; r < 8; ++r) {
            int row = rb + i * 16 + r;
#pragma unroll
            for (int t = 0; t < 4; ++t) {
                size_t o = (size_t)row * HIDDEN + cbl + t * 16;
                if (accumulate) out[o] += acc[i][t][r];
                else            out[o]  = acc[i][t][r];
            }
        }
    }
}

extern "C" void kernel_launch(void* const* d_in, const int* in_sizes, int n_in,
                              void* d_out, int out_size, void* d_ws, size_t ws_size,
                              hipStream_t stream) {
    const float* X  = (const float*)d_in[0];  // hidden_states  [NTOK, HIDDEN]
    const float* RW = (const float*)d_in[1];  // routing_weights [NTOK, E]
    const int*   RI = (const int*)  d_in[2];  // router_indices [NTOK, TOPK]
    const float* GU = (const float*)d_in[3];  // gate_up_proj [E, HIDDEN, 2*INTER]
    const float* DW = (const float*)d_in[4];  // down_proj    [E, INTER, HIDDEN]
    float* out = (float*)d_out;               // [NTOK, HIDDEN]

    // Workspace layout: combine f32 [NTOK][E], then inter bf16.
    const size_t combineBytes = (size_t)NTOK * NUM_EXPERTS * sizeof(float);
    const size_t interElems   = (size_t)NTOK * INTER;
    float* combine = (float*)d_ws;
    unsigned short* inter = (unsigned short*)((char*)d_ws + combineBytes);

    build_combine<<<(NTOK + 255) / 256, 256, 0, stream>>>(RW, RI, combine);

    const size_t needMerged = combineBytes + NUM_EXPERTS * interElems * sizeof(unsigned short);
    if (ws_size >= needMerged) {
        // Merged path: one pass-1 launch over all experts (grid.z = E), one
        // pass-2 launch accumulating all experts in registers. No out RMW.
        moe_gateup_swiglu<<<dim3(INTER / 64, NTOK / MT, NUM_EXPERTS), 256, 0, stream>>>(
            X, GU, combine, inter, 0);
        moe_down_combine<<<dim3(HIDDEN / 128, NTOK / MT), 256, 0, stream>>>(
            inter, DW, out, 0, NUM_EXPERTS, 0);
    } else {
        // Fallback: per-expert chain, single-expert inter buffer, out RMW.
        int n = NTOK * HIDDEN;
        zero_out_kernel<<<(n + 255) / 256, 256, 0, stream>>>(out, n);
        for (int e = 0; e < NUM_EXPERTS; ++e) {
            moe_gateup_swiglu<<<dim3(INTER / 64, NTOK / MT, 1), 256, 0, stream>>>(
                X, GU, combine, inter, e);
            moe_down_combine<<<dim3(HIDDEN / 128, NTOK / MT), 256, 0, stream>>>(
                inter, DW, out, e, 1, 1);
        }
    }
}
